// GridFMGPSTransformer_46729244180689
// MI455X (gfx1250) — matrix operand order repaired and checked by reading, not compiled, not run on hardware
//
#include <hip/hip_runtime.h>

typedef __attribute__((ext_vector_type(16))) __bf16 v16bf;
typedef __attribute__((ext_vector_type(8)))  __bf16 v8bf;
typedef __attribute__((ext_vector_type(8)))  float  v8f;

constexpr int NN = 2048, EE = 16384, LLAYERS = 8;
constexpr int IN_DIM = 9, HID = 256, PED = 20, OUT_DIM = 6, HEADS = 8, MASK_DIM = 6;
constexpr int ENCD = HID - PED;   // 236
constexpr int HD = HID / HEADS;   // 32

#define DEV static __device__ __forceinline__

DEV v8f wmma_bf16(v16bf a, v16bf b, v8f c) {
  return __builtin_amdgcn_wmma_f32_16x16x32_bf16(false, a, false, b, (short)0, c, false, false);
}

// A fragment: 16x32 (MxK) bf16, source row-major [16][lda].
// Layout: lanes 0-15 -> M=lane, K=0..15 in V0-7;
//         lanes 16-31 -> M=lane-16, K=16..31 in V0-7.
DEV v16bf load_frag_a(const __bf16* p, int lda) {
  int lane = threadIdx.x & 31;
  int m  = lane & 15;
  int kb = (lane >> 4) << 4;            // 0 or 16
  const __bf16* r = p + (size_t)m * lda;
  v8bf lo = *(const v8bf*)(r + kb);
  v8bf hi = *(const v8bf*)(r + kb + 8);
  v16bf a;
#pragma unroll
  for (int i = 0; i < 8; ++i) { a[i] = lo[i]; a[8 + i] = hi[i]; }
  return a;
}

// B fragment: 32x16 (KxN) bf16 from N-major storage [N][ldk] (i.e. W^T).
// Layout: lane&15 = N, lanes 0-15 hold K=0..15 in V0-7, lanes 16-31 hold K=16..31.
DEV v16bf load_frag_b_nmajor(const __bf16* p, int ldk) {
  int lane = threadIdx.x & 31;
  int n  = lane & 15;
  int kb = (lane >> 4) << 4;            // 0 or 16
  const __bf16* r = p + (size_t)n * ldk + kb;
  v8bf lo = *(const v8bf*)r;
  v8bf hi = *(const v8bf*)(r + 8);
  v16bf b;
#pragma unroll
  for (int i = 0; i < 8; ++i) { b[i] = lo[i]; b[8 + i] = hi[i]; }
  return b;
}

// B fragment: 32x16 (KxN) bf16 from K-major storage [K][ld] (strided per lane).
DEV v16bf load_frag_b_kmajor(const __bf16* p, int ld) {
  int lane = threadIdx.x & 31;
  int n  = lane & 15;
  int kb = (lane >> 4) << 4;
  v16bf b;
#pragma unroll
  for (int i = 0; i < 16; ++i) b[i] = p[(size_t)(kb + i) * ld + n];
  return b;
}

// ---------------- small scalar kernels ----------------

__global__ void zero_kernel(float* p, int n) {
  int i = blockIdx.x * blockDim.x + threadIdx.x;
  if (i < n) p[i] = 0.f;
}

// W[K][N] f32 -> Wt[N][K] bf16, batched over leading dim
__global__ void twt_kernel(const float* __restrict__ w, __bf16* __restrict__ wt,
                           int K, int N, long total) {
  long i = (long)blockIdx.x * blockDim.x + threadIdx.x;
  if (i >= total) return;
  long per = (long)K * N;
  long l = i / per, r = i % per;
  int k = (int)(r / N), n = (int)(r % N);
  wt[l * per + (long)n * K + k] = (__bf16)w[i];
}

__global__ void addbf_kernel(const float* a, const float* b, __bf16* o, int n) {
  int i = blockIdx.x * blockDim.x + threadIdx.x;
  if (i < n) o[i] = (__bf16)(a[i] + b[i]);
}

__global__ void addfbf_kernel(const float* a, const float* b, float* of, __bf16* ob, int n) {
  int i = blockIdx.x * blockDim.x + threadIdx.x;
  if (i < n) { float v = a[i] + b[i]; of[i] = v; ob[i] = (__bf16)v; }
}

// encoder: leaky(x_masked @ enc_w + enc_b), K=9 scalar
__global__ void enc_kernel(const float* __restrict__ x, const unsigned char* __restrict__ mask,
                           const float* __restrict__ mv, const float* __restrict__ w,
                           const float* __restrict__ bias, float* __restrict__ out) {
  int i = blockIdx.x * blockDim.x + threadIdx.x;
  if (i >= NN * ENCD) return;
  int n = i / ENCD, c = i % ENCD;
  float acc = bias[c];
#pragma unroll
  for (int j = 0; j < IN_DIM; ++j) {
    float xv = x[n * IN_DIM + j];
    if (j >= 3 && j < 3 + MASK_DIM && mask[n * MASK_DIM + (j - 3)]) xv = mv[j - 3];
    acc += xv * w[j * ENCD + c];
  }
  out[(size_t)n * ENCD + c] = acc > 0.f ? acc : 0.01f * acc;
}

// GINE: msg = relu(h[src] + edge_attr@ew + eb); atomic scatter-add into agg[dst]
__global__ void gine_msg_kernel(const float* __restrict__ h, const float* __restrict__ ea,
                                const int* __restrict__ src, const int* __restrict__ dst,
                                const float* __restrict__ ew, const float* __restrict__ eb,
                                float* __restrict__ agg) {
  int i = blockIdx.x * blockDim.x + threadIdx.x;
  if (i >= EE * HID) return;
  int e = i >> 8, c = i & 255;
  float em = ea[e * 2] * ew[c] + ea[e * 2 + 1] * ew[HID + c] + eb[c];
  float m = h[(size_t)src[e] * HID + c] + em;
  m = m > 0.f ? m : 0.f;
  atomicAdd(&agg[(size_t)dst[e] * HID + c], m);
}

// BatchNorm (training-mode batch stats, biased var): one block per column
__global__ void bn_kernel(const float* __restrict__ in, const float* __restrict__ add, int ldin,
                          const float* __restrict__ g, const float* __restrict__ b,
                          float* __restrict__ outf, __bf16* __restrict__ outb, int ldo, int rows) {
  __shared__ float s1[256], s2[256];
  int c = blockIdx.x, tid = threadIdx.x;
  float sum = 0.f, sq = 0.f;
  for (int r = tid; r < rows; r += 256) {
    float v = in[(size_t)r * ldin + c];
    if (add) v += add[(size_t)r * ldin + c];
    sum += v; sq += v * v;
  }
  s1[tid] = sum; s2[tid] = sq; __syncthreads();
  for (int s = 128; s > 0; s >>= 1) {
    if (tid < s) { s1[tid] += s1[tid + s]; s2[tid] += s2[tid + s]; }
    __syncthreads();
  }
  float mean = s1[0] / rows;
  float var  = s2[0] / rows - mean * mean;
  float sc = rsqrtf(var + 1e-5f) * g[c];
  float sh = b[c] - mean * sc;
  for (int r = tid; r < rows; r += 256) {
    float v = in[(size_t)r * ldin + c];
    if (add) v += add[(size_t)r * ldin + c];
    float o = v * sc + sh;
    if (outf) outf[(size_t)r * ldo + c] = o;
    if (outb) outb[(size_t)r * ldo + c] = (__bf16)o;
  }
}

// final decoder stage: d1 @ dec2w[256][6] + dec2b, scalar (N=6 < WMMA tile)
__global__ void dec2_kernel(const float* __restrict__ d1, const float* __restrict__ w,
                            const float* __restrict__ b, float* __restrict__ out) {
  int i = blockIdx.x * blockDim.x + threadIdx.x;
  if (i >= NN * OUT_DIM) return;
  int n = i / OUT_DIM, c = i % OUT_DIM;
  float acc = b[c];
  for (int k = 0; k < HID; ++k) acc += d1[(size_t)n * HID + k] * w[k * OUT_DIM + c];
  out[i] = acc;
}

// ---------------- WMMA GEMM: out = act(A[M,K] @ Wt^T + bias) ----------------
// A: bf16 [M][lda] row-major; Wt: bf16 [Nout][K] (pre-transposed).
// Register-blocked: one wave computes a 16x64 output strip (4 N-tiles),
// reusing one A fragment across 4 WMMAs per k-step (4 independent WMMAs
// also cover the bf16 WMMA->WMMA hazard distance).
__global__ void gemm_kernel(const __bf16* __restrict__ A, int lda,
                            const __bf16* __restrict__ Wt,
                            const float* __restrict__ bias,
                            float* __restrict__ outf, __bf16* __restrict__ outb, int ldo,
                            int groupsN, int K, int act, int totalWaves) {
  int wid = threadIdx.x >> 5;
  int w   = blockIdx.x * (blockDim.x >> 5) + wid;
  if (w >= totalWaves) return;
  int tm = w / groupsN, tg = w % groupsN;
  int lane = threadIdx.x & 31;
  const __bf16* Ap = A  + (size_t)tm * 16 * lda;
  const __bf16* Bp = Wt + (size_t)tg * 64 * K;
  v8f acc[4] = {};
  for (int k0 = 0; k0 < K; k0 += 32) {
    __builtin_prefetch(Ap + k0 + 32, 0, 1);   // -> global_prefetch_b8
    v16bf a  = load_frag_a(Ap + k0, lda);
    v16bf b0 = load_frag_b_nmajor(Bp + k0, K);
    v16bf b1 = load_frag_b_nmajor(Bp + (size_t)16 * K + k0, K);
    v16bf b2 = load_frag_b_nmajor(Bp + (size_t)32 * K + k0, K);
    v16bf b3 = load_frag_b_nmajor(Bp + (size_t)48 * K + k0, K);
    acc[0] = wmma_bf16(a, b0, acc[0]);
    acc[1] = wmma_bf16(a, b1, acc[1]);
    acc[2] = wmma_bf16(a, b2, acc[2]);
    acc[3] = wmma_bf16(a, b3, acc[3]);
  }
  int half = lane >> 4, col = lane & 15;
#pragma unroll
  for (int j = 0; j < 4; ++j) {
    int gcol = (tg * 4 + j) * 16 + col;
    float bb = bias[gcol];
#pragma unroll
    for (int g = 0; g < 8; ++g) {
      int grow = tm * 16 + g + 8 * half;
      float v = acc[j][g] + bb;
      if (act == 1) v = v > 0.f ? v : 0.f;
      else if (act == 2) v = v > 0.f ? v : 0.01f * v;
      if (outf) outf[(size_t)grow * ldo + gcol] = v;
      if (outb) outb[(size_t)grow * ldo + gcol] = (__bf16)v;
    }
  }
}

// ---------------- flash attention (per wave: one 16-row query tile x one head) ----
// qkv: bf16 [N][768] = [q(256)|k(256)|v(256)], head h occupies cols h*32..h*32+31.
// S tile = ONE wmma (K==HD==32); online softmax; P restaged through LDS into
// A-fragment layout; V chunk staged into LDS with coalesced b128 loads
// (lane = key row), then fragment-gathered from LDS; O = two wmmas / 32-key chunk.
__global__ void attn_kernel(const __bf16* __restrict__ qkv, __bf16* __restrict__ obf) {
  __shared__ alignas(16) __bf16 lp[4][16 * 32];   // P tiles, per wave
  __shared__ alignas(16) __bf16 lv[4][32 * 32];   // V chunks, per wave
  int wid  = threadIdx.x >> 5;
  int lane = threadIdx.x & 31;
  int t = blockIdx.x * 4 + wid;
  int rt = t >> 3, head = t & 7;
  int row0 = rt * 16;
  const float scale = 0.17677669529663687f;   // 1/sqrt(32)
  int half = lane >> 4, nl = lane & 15;

  v16bf aq = load_frag_a(qkv + (size_t)row0 * 768 + head * HD, 768);
  v8f o0 = {}, o1 = {};
  float m_run[8], l_run[8];
#pragma unroll
  for (int g = 0; g < 8; ++g) { m_run[g] = -1e30f; l_run[g] = 0.f; }
  __bf16* myp = lp[wid];
  __bf16* myv = lv[wid];

  for (int c = 0; c < NN; c += 32) {
    const __bf16* kp = qkv + (size_t)c * 768 + HID + head * HD;
    v16bf bk0 = load_frag_b_nmajor(kp, 768);
    v16bf bk1 = load_frag_b_nmajor(kp + (size_t)16 * 768, 768);
    v8f z = {};
    v8f s0 = wmma_bf16(aq, bk0, z);
    v8f s1 = wmma_bf16(aq, bk1, z);
    // stage V chunk (32 keys x 32 dims): lane = key, 4x b128 per lane
    {
      const v8bf* vsrc = (const v8bf*)(qkv + (size_t)(c + lane) * 768 + 2 * HID + head * HD);
      v8bf* vdst = (v8bf*)(myv + lane * 32);
      vdst[0] = vsrc[0]; vdst[1] = vsrc[1]; vdst[2] = vsrc[2]; vdst[3] = vsrc[3];
    }
    __syncthreads();                       // prior P/V tiles fully consumed
#pragma unroll
    for (int g = 0; g < 8; ++g) {
      float x0 = s0[g] * scale, x1 = s1[g] * scale;
      float mx = fmaxf(x0, x1);
      mx = fmaxf(mx, __shfl_xor(mx, 1, 32));
      mx = fmaxf(mx, __shfl_xor(mx, 2, 32));
      mx = fmaxf(mx, __shfl_xor(mx, 4, 32));
      mx = fmaxf(mx, __shfl_xor(mx, 8, 32));
      float mnew = fmaxf(m_run[g], mx);
      float cf = __expf(m_run[g] - mnew);
      float p0 = __expf(x0 - mnew), p1 = __expf(x1 - mnew);
      float rs = p0 + p1;
      rs += __shfl_xor(rs, 1, 32);
      rs += __shfl_xor(rs, 2, 32);
      rs += __shfl_xor(rs, 4, 32);
      rs += __shfl_xor(rs, 8, 32);
      l_run[g] = l_run[g] * cf + rs;
      m_run[g] = mnew;
      o0[g] *= cf; o1[g] *= cf;
      int row = g + 8 * half;              // C-layout position of (x0,x1)
      myp[row * 32 + nl]      = (__bf16)p0;
      myp[row * 32 + 16 + nl] = (__bf16)p1;
    }
    __syncthreads();                       // P + V tiles visible
    v16bf ap  = load_frag_a(myp, 32);
    v16bf bv0 = load_frag_b_kmajor(myv, 32);
    v16bf bv1 = load_frag_b_kmajor(myv + 16, 32);
    o0 = wmma_bf16(ap, bv0, o0);
    o1 = wmma_bf16(ap, bv1, o1);
  }
#pragma unroll
  for (int g = 0; g < 8; ++g) {
    float inv = 1.f / l_run[g];
    int row = row0 + g + 8 * half;
    obf[(size_t)row * HID + head * HD + nl]      = (__bf16)(o0[g] * inv);
    obf[(size_t)row * HID + head * HD + 16 + nl] = (__bf16)(o1[g] * inv);
  }
}

// ---------------- host orchestration ----------------

extern "C" void kernel_launch(void* const* d_in, const int* in_sizes, int n_in,
                              void* d_out, int out_size, void* d_ws, size_t ws_size,
                              hipStream_t stream) {
  (void)in_sizes; (void)n_in; (void)out_size; (void)ws_size;
  const float* x          = (const float*)d_in[0];
  const float* pe         = (const float*)d_in[1];
  const float* edge_attr  = (const float*)d_in[2];
  const float* mask_value = (const float*)d_in[3];
  const float* enc_w = (const float*)d_in[4];
  const float* enc_b = (const float*)d_in[5];
  const float* in_g  = (const float*)d_in[6];
  const float* in_b  = (const float*)d_in[7];
  const float* pe_g  = (const float*)d_in[8];
  const float* pe_b  = (const float*)d_in[9];
  const float* gine_ew = (const float*)d_in[10];
  const float* gine_eb = (const float*)d_in[11];
  const float* gine_mw = (const float*)d_in[12];
  const float* gine_mb = (const float*)d_in[13];
  const float* attn_inw  = (const float*)d_in[14];
  const float* attn_inb  = (const float*)d_in[15];
  const float* attn_outw = (const float*)d_in[16];
  const float* attn_outb = (const float*)d_in[17];
  const float* mlp1w = (const float*)d_in[18];
  const float* mlp1b = (const float*)d_in[19];
  const float* mlp2w = (const float*)d_in[20];
  const float* mlp2b = (const float*)d_in[21];
  const float* n1g = (const float*)d_in[22];
  const float* n1b = (const float*)d_in[23];
  const float* n2g = (const float*)d_in[24];
  const float* n2b = (const float*)d_in[25];
  const float* n3g = (const float*)d_in[26];
  const float* n3b = (const float*)d_in[27];
  const float* obn_g = (const float*)d_in[28];
  const float* obn_b = (const float*)d_in[29];
  const float* pd_g = (const float*)d_in[30];
  const float* pd_b = (const float*)d_in[31];
  const float* dec1w = (const float*)d_in[32];
  const float* dec1b = (const float*)d_in[33];
  const float* dec2w = (const float*)d_in[34];
  const float* dec2b = (const float*)d_in[35];
  const int* edge_index = (const int*)d_in[36];
  const unsigned char* mask = (const unsigned char*)d_in[37];
  const int* src = edge_index;
  const int* dst = edge_index + EE;

  char* ws = (char*)d_ws;
  size_t off = 0;
  auto alloc = [&](size_t bytes) { void* p = ws + off; off += (bytes + 255) & ~(size_t)255; return p; };

  const size_t NH = (size_t)NN * HID;
  float*  encv   = (float*)alloc((size_t)NN * ENCD * 4);
  float*  h      = (float*)alloc(NH * 4);
  __bf16* h_bf   = (__bf16*)alloc(NH * 2);
  float*  agg    = (float*)alloc(NH * 4);
  __bf16* tmpbf  = (__bf16*)alloc(NH * 2);
  float*  hl     = (float*)alloc(NH * 4);
  float*  hln    = (float*)alloc(NH * 4);
  __bf16* qkv_bf = (__bf16*)alloc((size_t)NN * 768 * 2);
  __bf16* o_bf   = (__bf16*)alloc(NH * 2);
  float*  attn   = (float*)alloc(NH * 4);
  float*  ha     = (float*)alloc(NH * 4);
  float*  outf   = (float*)alloc(NH * 4);
  __bf16* out_bf = (__bf16*)alloc(NH * 2);
  __bf16* t1_bf  = (__bf16*)alloc((size_t)NN * 512 * 2);
  float*  t2     = (float*)alloc(NH * 4);
  float*  bn3    = (float*)alloc(NH * 4);
  float*  hpd    = (float*)alloc(NH * 4);
  __bf16* hpd_bf = (__bf16*)alloc(NH * 2);
  float*  d1f    = (float*)alloc(NH * 4);
  // transposed bf16 weights
  __bf16* mwt   = (__bf16*)alloc((size_t)LLAYERS * 256 * 256 * 2);
  __bf16* inwt  = (__bf16*)alloc((size_t)LLAYERS * 256 * 768 * 2);
  __bf16* outwt = (__bf16*)alloc((size_t)LLAYERS * 256 * 256 * 2);
  __bf16* m1wt  = (__bf16*)alloc((size_t)LLAYERS * 256 * 512 * 2);
  __bf16* m2wt  = (__bf16*)alloc((size_t)LLAYERS * 512 * 256 * 2);
  __bf16* d1wt  = (__bf16*)alloc((size_t)256 * 256 * 2);

  auto cdiv = [](long a, long b) { return (int)((a + b - 1) / b); };
  auto twt = [&](const float* w, __bf16* wt, int K, int N, int Lb) {
    long total = (long)Lb * K * N;
    twt_kernel<<<cdiv(total, 256), 256, 0, stream>>>(w, wt, K, N, total);
  };
  auto gemm = [&](const __bf16* A, int lda, const __bf16* Wt, const float* bias,
                  float* of, __bf16* ob, int Nout, int K, int act) {
    int groupsN = Nout / 64;                    // 16x64 strip per wave
    int totalWaves = (NN / 16) * groupsN;
    gemm_kernel<<<totalWaves / 4, 128, 0, stream>>>(A, lda, Wt, bias, of, ob, Nout,
                                                    groupsN, K, act, totalWaves);
  };
  auto bn = [&](const float* in, const float* add, int ldin, const float* g, const float* b,
                float* of, __bf16* ob, int ldo, int cols) {
    bn_kernel<<<cols, 256, 0, stream>>>(in, add, ldin, g, b, of, ob, ldo, NN);
  };

  // weight conversion + transpose (deterministic, every call)
  twt(gine_mw,  mwt,  256, 256, LLAYERS);
  twt(attn_inw, inwt, 256, 768, LLAYERS);
  twt(attn_outw, outwt, 256, 256, LLAYERS);
  twt(mlp1w, m1wt, 256, 512, LLAYERS);
  twt(mlp2w, m2wt, 512, 256, LLAYERS);
  twt(dec1w, d1wt, 256, 256, 1);

  // encoder + input norms -> h[:, :236] | BN(pe) -> h[:, 236:256]
  enc_kernel<<<cdiv((long)NN * ENCD, 256), 256, 0, stream>>>(x, mask, mask_value, enc_w, enc_b, encv);
  bn(encv, nullptr, ENCD, in_g, in_b, h, h_bf, HID, ENCD);
  bn(pe,   nullptr, PED,  pe_g, pe_b, h + ENCD, h_bf + ENCD, HID, PED);

  for (int l = 0; l < LLAYERS; ++l) {
    // --- GINE local branch ---
    zero_kernel<<<cdiv(NH, 256), 256, 0, stream>>>(agg, (int)NH);
    gine_msg_kernel<<<cdiv((long)EE * HID, 256), 256, 0, stream>>>(
        h, edge_attr, src, dst, gine_ew + l * 2 * HID, gine_eb + l * HID, agg);
    addbf_kernel<<<cdiv(NH, 256), 256, 0, stream>>>(h, agg, tmpbf, (int)NH);
    gemm(tmpbf, HID, mwt + (size_t)l * 256 * 256, gine_mb + l * HID, hl, nullptr, HID, HID, 2);
    bn(hl, h, HID, n1g + l * HID, n1b + l * HID, hln, nullptr, HID, HID);
    // --- global attention branch ---
    gemm(h_bf, HID, inwt + (size_t)l * 256 * 768, attn_inb + l * 768, nullptr, qkv_bf, 768, HID, 0);
    attn_kernel<<<(NN / 16) * HEADS / 4, 128, 0, stream>>>(qkv_bf, o_bf);
    gemm(o_bf, HID, outwt + (size_t)l * 256 * 256, attn_outb + l * HID, attn, nullptr, HID, HID, 0);
    bn(attn, h, HID, n2g + l * HID, n2b + l * HID, ha, nullptr, HID, HID);
    // --- combine + FFN ---
    addfbf_kernel<<<cdiv(NH, 256), 256, 0, stream>>>(hln, ha, outf, out_bf, (int)NH);
    gemm(out_bf, HID, m1wt + (size_t)l * 256 * 512, mlp1b + l * 512, nullptr, t1_bf, 512, HID, 1);
    gemm(t1_bf, 512, m2wt + (size_t)l * 512 * 256, mlp2b + l * HID, t2, nullptr, HID, 512, 0);
    bn(t2, outf, HID, n3g + l * HID, n3b + l * HID, bn3, nullptr, HID, HID);
    bn(bn3, nullptr, HID, obn_g + l * HID, obn_b + l * HID, h, h_bf, HID, HID);
  }

  // post norm + decoder
  bn(h, nullptr, HID, pd_g, pd_b, hpd, hpd_bf, HID, HID);
  gemm(hpd_bf, HID, d1wt, dec1b, d1f, nullptr, HID, HID, 2);
  dec2_kernel<<<cdiv((long)NN * OUT_DIM, 256), 256, 0, stream>>>(d1f, dec2w, dec2b, (float*)d_out);
}
